// StackEncoder_35725537968238
// MI455X (gfx1250) — matrix operand order repaired and checked by reading, not compiled, not run on hardware
//
#include <hip/hip_runtime.h>

// SPINN stack encoder, B=64, L=48, D=256, fixed schedule (48 shifts, 48 reduces)
// => left-chain TreeLSTM.
// Phase 1: big parallel left-child gate GEMM (WMMA bf16, whole chip).
// Phase 2: 48-step sequential recurrence on one WGP:
//   - running h in LDS (bf16, WMMA A source), running c in VGPRs
//   - per-step left-contribution slice (64 x 1024 f32) staged into LDS by the
//     Tensor Data Mover (tensor_load_to_lds), overlapped with the WMMA K-loop.

typedef __bf16 bf16_t;
typedef __attribute__((ext_vector_type(16))) __bf16 v16bf;
typedef __attribute__((ext_vector_type(8)))  __bf16 v8bf;
typedef __attribute__((ext_vector_type(8)))  float  v8f;
typedef __attribute__((ext_vector_type(4)))  unsigned v4u;
typedef __attribute__((ext_vector_type(8)))  int      v8i;
typedef __attribute__((ext_vector_type(4)))  int      v4i;

#define B_DIM 64
#define L_DIM 48
#define D_DIM 256
#define ROWS  (B_DIM * L_DIM)   // 3072
#define GCOLS 1024              // 4 gates * 256

__device__ __forceinline__ float sigmoidf_(float x) {
    return 1.0f / (1.0f + __expf(-x));
}

// A-fragment (16x32 bf16, MxK), contiguous k order:
// lane lid(0..15), half = lane>>4 ; elements 0..15 = K[k0+half*16 .. +15].
// base must point at row + k0 + half*16.
__device__ __forceinline__ v16bf load_a_frag(const bf16_t* base) {
    v8bf lo = *(const v8bf*)(base);
    v8bf hi = *(const v8bf*)(base + 8);
    v16bf a;
#pragma unroll
    for (int j = 0; j < 8; ++j) { a[j] = lo[j]; a[8 + j] = hi[j]; }
    return a;
}

// TDM: DMA one step's left-contribution slice Lout[(b*48+lt), 0:1024] for all b
// into LDS as a 2D tile: 64 rows x 1024 f32, global row stride 48*1024 elements.
__device__ __forceinline__ void tdm_load_slice(const float* gsrc, unsigned lds_off) {
    const unsigned long long ga = (unsigned long long)(size_t)gsrc;
    // D# group 0: count=1 | lds_addr | global_addr[56:0] | type=2 (bits 127:126)
    v4u g0 = { 1u,
               lds_off,
               (unsigned)(ga & 0xFFFFFFFFu),
               (unsigned)((ga >> 32) & 0x01FFFFFFu) | 0x80000000u };
    // D# group 1:
    //  w0: workgroup_mask=0, data_size=2 (4B) at bits 17:16
    //  w1[31:16] = tensor_dim0[15:0] = 1024
    //  w2[31:16] = tensor_dim1[15:0] = 64
    //  w3[31:16] = tile_dim0 = 1024
    //  w4[15:0]  = tile_dim1 = 64
    //  w5        = tensor_dim0_stride[31:0] = 48*1024 = 49152 (data_size units)
    v8i g1 = { (int)0x20000,
               (int)(1024u << 16),
               (int)(64u << 16),
               (int)(1024u << 16),
               64,
               49152,
               0,
               0 };
    v4i z4 = { 0, 0, 0, 0 };
#if defined(__clang_major__) && (__clang_major__ >= 23)
    v8i z8 = { 0, 0, 0, 0, 0, 0, 0, 0 };
    __builtin_amdgcn_tensor_load_to_lds(g0, g1, z4, z4, z8, 0);
#else
    __builtin_amdgcn_tensor_load_to_lds(g0, g1, z4, z4, 0);
#endif
}

// -------------------- prep: f32 -> bf16 --------------------
__global__ void f32_to_bf16_kernel(const float* __restrict__ src,
                                   bf16_t* __restrict__ dst, int n) {
    int i = blockIdx.x * blockDim.x + threadIdx.x;
    if (i < n) dst[i] = (bf16_t)src[i];
}

// -------------------- phase 1: left-child gate GEMM --------------------
__global__ void __launch_bounds__(256)
left_gemm_kernel(const bf16_t* __restrict__ Abf,   // 3072 x 256 bf16
                 const bf16_t* __restrict__ Wil, const bf16_t* __restrict__ Wol,
                 const bf16_t* __restrict__ Wfl, const bf16_t* __restrict__ Wul,
                 const float* __restrict__ bi,  const float* __restrict__ bo,
                 const float* __restrict__ bfv, const float* __restrict__ bu,
                 float* __restrict__ Lout)         // 3072 x 1024 f32
{
    const int lane = threadIdx.x & 31;
    const int widx = threadIdx.x >> 5;          // 8 waves / block
    const int w    = blockIdx.x * 8 + widx;     // 0 .. 12287
    const int mt   = w >> 6;                    // 192 row tiles
    const int nt   = w & 63;                    // 64 col tiles
    const int lid  = lane & 15;
    const int half = lane >> 4;

    const int gate = nt >> 4;
    const int ncol = ((nt & 15) << 4) + lid;    // 0..255 within gate
    const bf16_t* Wg = (gate == 0) ? Wil : (gate == 1) ? Wol : (gate == 2) ? Wfl : Wul;
    const float*  bg = (gate == 0) ? bi  : (gate == 1) ? bo  : (gate == 2) ? bfv : bu;

    const bf16_t* arow = Abf + (size_t)(mt * 16 + lid) * D_DIM;
    const bf16_t* brow = Wg  + (size_t)ncol * D_DIM;    // col n of B == row n of U

    v8f acc = {};
#pragma unroll
    for (int k0 = 0; k0 < D_DIM; k0 += 32) {
        v16bf a = load_a_frag(arow + k0 + half * 16);
        v16bf b = *(const v16bf*)(brow + k0 + half * 16);
        acc = __builtin_amdgcn_wmma_f32_16x16x32_bf16(false, a, false, b,
                                                      (short)0, acc, false, false);
    }

    const float bias  = bg[ncol];
    const int   ncolg = nt * 16 + lid;          // global col 0..1023
#pragma unroll
    for (int r = 0; r < 8; ++r) {
        const int row = mt * 16 + r + half * 8; // C/D: VGPR r -> M = r + half*8
        Lout[(size_t)row * GCOLS + ncolg] = acc[r] + bias;
    }
}

// -------------------- phase 2: 48-step recurrence, single WGP --------------------
__global__ void __launch_bounds__(1024)
recurrent_kernel(const bf16_t* __restrict__ th_bf,    // (B*L) x 256 bf16 tokens_h
                 const float*  __restrict__ tokens_c, // B x L x 256 f32
                 const float*  __restrict__ Lout,     // (B*L) x 1024 left pre-acts (+bias)
                 const bf16_t* __restrict__ Wil, const bf16_t* __restrict__ Wor,
                 const bf16_t* __restrict__ Wfr, const bf16_t* __restrict__ Wur,
                 float* __restrict__ out)             // B x 256 f32
{
    extern __shared__ char smem[];
    bf16_t* hbuf = (bf16_t*)smem;                       // 64*256 bf16 = 32 KB
    float*  lbuf = (float*)(smem + 32 * 1024);          // 64*1024 f32 = 256 KB
    const unsigned lbuf_off = (unsigned)(size_t)(void*)lbuf;  // LDS byte offset

    const int tid  = threadIdx.x;
    const int lane = tid & 31;
    const int widx = tid >> 5;              // 32 waves
    const int lid  = lane & 15;
    const int half = lane >> 4;

    // prime the TDM pipeline: stage slice for step 0 (lt = 46)
    if (widx == 0) {
        tdm_load_slice(Lout + (size_t)46 * GCOLS, lbuf_off);
    }

    // init h = tokens_h[:, 47]
    for (int i = tid; i < B_DIM * D_DIM; i += 1024) {
        const int b = i >> 8, d = i & 255;
        hbuf[i] = th_bf[(size_t)(b * L_DIM + (L_DIM - 1)) * D_DIM + d];
    }

    // each wave owns 2 of the 64 (mt,nt) output tiles; its c-state lives in VGPRs
    v8f c_reg[2];
#pragma unroll
    for (int t = 0; t < 2; ++t) {
        const int tg = widx + t * 32;
        const int mt = tg >> 4, nt = tg & 15;
        const int d  = nt * 16 + lid;
#pragma unroll
        for (int r = 0; r < 8; ++r) {
            const int b = mt * 16 + r + half * 8;
            c_reg[t][r] = tokens_c[(size_t)(b * L_DIM + (L_DIM - 1)) * D_DIM + d];
        }
    }
    __syncthreads();

    v8f accI[2], accO[2], accF[2], accU[2];
    float hnew[2][8];

    for (int s = 0; s < 48; ++s) {
        const int lt = (s < 47) ? (46 - s) : 0;

        // ---- K-loop: right-child gate WMMAs (reads hbuf; TDM fills lbuf async)
#pragma unroll
        for (int t = 0; t < 2; ++t) {
            const int tg = widx + t * 32;
            const int mt = tg >> 4, nt = tg & 15;
            const int d  = nt * 16 + lid;

            v8f ai = {}, ao = {}, af = {}, au = {};
            const bf16_t* arow = hbuf + (mt * 16 + lid) * D_DIM;
            const size_t  boff = (size_t)d * D_DIM;
#pragma unroll
            for (int k0 = 0; k0 < D_DIM; k0 += 32) {
                v16bf a  = load_a_frag(arow + k0 + half * 16);
                v16bf b0 = *(const v16bf*)(Wil + boff + k0 + half * 16);
                ai = __builtin_amdgcn_wmma_f32_16x16x32_bf16(false, a, false, b0,
                                                             (short)0, ai, false, false);
                v16bf b1 = *(const v16bf*)(Wor + boff + k0 + half * 16);
                ao = __builtin_amdgcn_wmma_f32_16x16x32_bf16(false, a, false, b1,
                                                             (short)0, ao, false, false);
                v16bf b2 = *(const v16bf*)(Wfr + boff + k0 + half * 16);
                af = __builtin_amdgcn_wmma_f32_16x16x32_bf16(false, a, false, b2,
                                                             (short)0, af, false, false);
                v16bf b3 = *(const v16bf*)(Wur + boff + k0 + half * 16);
                au = __builtin_amdgcn_wmma_f32_16x16x32_bf16(false, a, false, b3,
                                                             (short)0, au, false, false);
            }
            accI[t] = ai; accO[t] = ao; accF[t] = af; accU[t] = au;
        }

        // TDM slice for this step (issued last step / primed): retire + publish
        if (widx == 0) {
            __builtin_amdgcn_s_wait_tensorcnt((short)0);
        }
        __syncthreads();   // hbuf read phase closed; lbuf valid for all waves

        // ---- fixup: gates = WMMA acc + staged left pre-acts (LDS) ; update c,h
#pragma unroll
        for (int t = 0; t < 2; ++t) {
            const int tg = widx + t * 32;
            const int mt = tg >> 4, nt = tg & 15;
            const int d  = nt * 16 + lid;
#pragma unroll
            for (int r = 0; r < 8; ++r) {
                const int b   = mt * 16 + r + half * 8;
                const int lrb = b * GCOLS + d;
                const float i_g = accI[t][r] + lbuf[lrb + 0 * 256];
                const float o_g = accO[t][r] + lbuf[lrb + 1 * 256];
                const float f_g = accF[t][r] + lbuf[lrb + 2 * 256];
                const float u_g = accU[t][r] + lbuf[lrb + 3 * 256];
                const size_t crow = (size_t)(b * L_DIM + lt);
                const float c_l = tokens_c[crow * D_DIM + d];
                const float c   = sigmoidf_(i_g) * tanhf(u_g)
                                + sigmoidf_(f_g) * (c_l + c_reg[t][r]);
                c_reg[t][r] = c;
                hnew[t][r]  = o_g * tanhf(c);   // o NOT sigmoided (ref quirk)
            }
        }
        __syncthreads();   // all lbuf reads done -> TDM may overwrite

        // ---- publish new h ; kick TDM for next step's slice
#pragma unroll
        for (int t = 0; t < 2; ++t) {
            const int tg = widx + t * 32;
            const int mt = tg >> 4, nt = tg & 15;
            const int d  = nt * 16 + lid;
#pragma unroll
            for (int r = 0; r < 8; ++r) {
                const int b = mt * 16 + r + half * 8;
                hbuf[b * D_DIM + d] = (bf16_t)hnew[t][r];
                if (s == 47)
                    out[(size_t)b * D_DIM + d] = hnew[t][r];
            }
        }
        if (widx == 0 && s < 47) {
            const int ltn = (s + 1 < 47) ? (45 - s) : 0;
            tdm_load_slice(Lout + (size_t)ltn * GCOLS, lbuf_off);
            __builtin_prefetch(tokens_c + (size_t)ltn * D_DIM, 0, 1);
        }
        __syncthreads();   // hbuf writes visible before next step's reads
    }
}

// -------------------- host --------------------
extern "C" void kernel_launch(void* const* d_in, const int* in_sizes, int n_in,
                              void* d_out, int out_size, void* d_ws, size_t ws_size,
                              hipStream_t stream) {
    const float* tokens_h = (const float*)d_in[0];
    const float* tokens_c = (const float*)d_in[1];
    // d_in[2] = transitions: fixed shift^48 reduce^48 pattern, folded into code.
    const float* Usrc[7] = {
        (const float*)d_in[3],  // U_il
        (const float*)d_in[4],  // U_ol
        (const float*)d_in[5],  // U_or
        (const float*)d_in[6],  // U_fl
        (const float*)d_in[7],  // U_fr
        (const float*)d_in[8],  // U_ul
        (const float*)d_in[9],  // U_ur
    };
    const float* b_i = (const float*)d_in[10];
    const float* b_o = (const float*)d_in[11];
    const float* b_f = (const float*)d_in[12];
    const float* b_u = (const float*)d_in[13];

    char* ws = (char*)d_ws;
    size_t off = 0;
    auto wsalloc = [&](size_t bytes) -> void* {
        void* p = ws + off;
        off = (off + bytes + 255) & ~(size_t)255;
        return p;
    };

    bf16_t* th_bf = (bf16_t*)wsalloc((size_t)ROWS * D_DIM * sizeof(bf16_t));
    bf16_t* Ubf[7];
    for (int j = 0; j < 7; ++j)
        Ubf[j] = (bf16_t*)wsalloc((size_t)D_DIM * D_DIM * sizeof(bf16_t));
    float* Lout = (float*)wsalloc((size_t)ROWS * GCOLS * sizeof(float));

    {
        const int n = ROWS * D_DIM;
        f32_to_bf16_kernel<<<(n + 255) / 256, 256, 0, stream>>>(tokens_h, th_bf, n);
        const int m = D_DIM * D_DIM;
        for (int j = 0; j < 7; ++j)
            f32_to_bf16_kernel<<<(m + 255) / 256, 256, 0, stream>>>(Usrc[j], Ubf[j], m);
    }

    left_gemm_kernel<<<(ROWS / 16) * (GCOLS / 16) / 8, 256, 0, stream>>>(
        th_bf, Ubf[0], Ubf[1], Ubf[3], Ubf[5], b_i, b_o, b_f, b_u, Lout);

    // phase 2: one WGP, 288 KB dynamic LDS (32 KB h + 256 KB TDM-staged gates)
    recurrent_kernel<<<1, 1024, 288 * 1024, stream>>>(
        th_bf, tokens_c, Lout, Ubf[0], Ubf[2], Ubf[4], Ubf[6], (float*)d_out);
}
